// DeformableNeighborhoodAttention_25649544691723
// MI455X (gfx1250) — matrix operand order repaired and checked by reading, not compiled, not run on hardware
//
#include <hip/hip_runtime.h>
#include <math.h>

// ---------------- problem constants (from reference) ----------------
#define B_     4
#define C_     128
#define H_     56
#define W_     56
#define HW_    (H_ * W_)          // 3136
#define HEADS_ 4
#define HD_    32
#define GC_    32                 // group channels == head dim
#define K_     7
#define P_     49
#define PAD_   3
#define NPIX   (B_ * HW_)         // 12544
#define SCALE_ 0.17677669529663687f   // 32^-0.5
#define EPS_   1e-6f

#define LDS_CSTRIDE 20            // 16 pixels + 4 pad floats: halves hit disjoint banks

typedef _Float16 half16 __attribute__((ext_vector_type(16)));
typedef float    float8 __attribute__((ext_vector_type(8)));

// =====================================================================
// conv1x1 as GEMM:  Y[b, o, hw] = sum_c X[b, c, hw] * Wt[o, c] + bias[o]
//
// One wave32 -> one 16-pixel M tile x ALL 128 out-channels (8 N tiles,
// 8 f32 accumulators, 32 static v_wmma_f32_16x16x32_f16).
//
// A tile (16 pixels x 128 channels, f32) is staged into LDS with
// GLOBAL_LOAD_ASYNC_TO_LDS_B128 (ASYNCcnt-tracked, coalesced 16B
// transfers, no VGPR staging), completed with s_wait_asynccnt 0.
// The wave consumes only its own LDS region -> no workgroup barrier.
//
// Register packing (16-bit A 16x32): lane -> M,
//   a[e]=K(half*16+e).
// B (32x16): lane -> N, b[e]=K(half*16+e).
// D (f32 16x16): lane -> N, element r -> M = r + half*8.
// =====================================================================
__global__ __launch_bounds__(128)
void gemm_conv1x1_wmma(const float* __restrict__ X, const float* __restrict__ Wt,
                       const float* __restrict__ bias, float* __restrict__ Y) {
    __shared__ float atile[4][C_ * LDS_CSTRIDE];   // 4 waves x 10240B = 40KB

    const int lane  = threadIdx.x & 31;
    const int wv    = threadIdx.x >> 5;            // wave in block (0..3)
    const int mTile = blockIdx.x * 4 + wv;         // 784 M tiles total
    if (mTile >= NPIX / 16) return;                // uniform per wave
    const int m0   = mTile * 16;
    const int b    = m0 / HW_;                     // HW_ % 16 == 0: no batch straddle
    const int hw0  = m0 % HW_;
    const int half = lane >> 4;
    const int l16  = lane & 15;

    const float* Ag = X + (size_t)b * C_ * HW_ + hw0;      // + c*HW_ + pixel
    float* As = atile[wv];
    const uint32_t ldsBase = (uint32_t)(uintptr_t)As;

    // ---- async stage: 512 x b128 (4 floats) = 16 pixels x 128 channels ----
#pragma unroll
    for (int i = 0; i < 16; ++i) {
        int j  = i * 32 + lane;          // 0..511
        int c  = j >> 2;                 // channel 0..127
        int ch = (j & 3) * 4;            // pixel chunk 0,4,8,12
        uint64_t ga = (uint64_t)(uintptr_t)(Ag + (size_t)c * HW_ + ch);
        uint32_t ld = ldsBase + (uint32_t)(c * LDS_CSTRIDE + ch) * 4u;
        asm volatile("global_load_async_to_lds_b128 %0, %1, off"
                     :: "v"(ld), "v"(ga) : "memory");
    }
    asm volatile("s_wait_asynccnt 0" ::: "memory");   // wave-private tile: no barrier

    float8 acc[8];
#pragma unroll
    for (int nt = 0; nt < 8; ++nt)
        acc[nt] = float8{0.f, 0.f, 0.f, 0.f, 0.f, 0.f, 0.f, 0.f};

#pragma unroll
    for (int kb = 0; kb < C_; kb += 32) {
        // A fragment from LDS (ds_load_b32, conflict-free across halves)
        half16 a;
#pragma unroll
        for (int t = 0; t < 8; ++t) {
            a[t]     = (_Float16)As[(kb + half * 16 + t) * LDS_CSTRIDE + l16];
            a[8 + t] = (_Float16)As[(kb + half * 16 + 8 + t) * LDS_CSTRIDE + l16];
        }
        if (kb + 32 < C_) {
            __builtin_prefetch(Wt + (size_t)l16 * C_ + kb + 32, 0, 1); // global_prefetch_b8
        }
#pragma unroll
        for (int nt = 0; nt < 8; ++nt) {
            const float4* wvp = reinterpret_cast<const float4*>(
                Wt + (size_t)(nt * 16 + l16) * C_ + kb + half * 16);
            half16 bm;
#pragma unroll
            for (int jj = 0; jj < 4; ++jj) {
                float4 w4 = wvp[jj];
                bm[4 * jj + 0] = (_Float16)w4.x;
                bm[4 * jj + 1] = (_Float16)w4.y;
                bm[4 * jj + 2] = (_Float16)w4.z;
                bm[4 * jj + 3] = (_Float16)w4.w;
            }
            acc[nt] = __builtin_amdgcn_wmma_f32_16x16x32_f16(false, a, false, bm,
                                                             (short)0, acc[nt],
                                                             false, false);
        }
    }

#pragma unroll
    for (int nt = 0; nt < 8; ++nt) {
        const int o = nt * 16 + l16;
        const float bv = bias[o];
        float* Yb = Y + (size_t)b * C_ * HW_ + (size_t)o * HW_ + hw0 + half * 8;
#pragma unroll
        for (int r = 0; r < 8; ++r) Yb[r] = acc[nt][r] + bv;
    }
}

// =====================================================================
// Offset branch + bilinear grid sample.
// One wave per (batch*group, pixel); lane = group channel (GC_ == 32).
// dwconv7x7 -> LN(32ch, butterfly reduce) -> exact GELU -> pointwise(2) ->
// tanh/range -> pos = off + ref -> bilinear sample of x (zero pad,
// align_corners=True semantics, mirrors reference corner()).
// =====================================================================
__global__ __launch_bounds__(256)
void offset_sample(const float* __restrict__ X, const float* __restrict__ Q,
                   const float* __restrict__ dw_w, const float* __restrict__ dw_b,
                   const float* __restrict__ ln_w, const float* __restrict__ ln_b,
                   const float* __restrict__ pw_w, float* __restrict__ Xs) {
    const int lane = threadIdx.x & 31;
    const int wave = blockIdx.x * 8 + (threadIdx.x >> 5);
    if (wave >= B_ * HEADS_ * HW_) return;
    const int hw = wave % HW_;
    const int bg = wave / HW_;
    const int g  = bg % HEADS_;
    const int b  = bg / HEADS_;
    const int y  = hw / W_, x = hw % W_;
    const int cg = g * GC_ + lane;

    const float* qc = Q + ((size_t)b * C_ + cg) * HW_;

    // depthwise 7x7 (zero pad); (y,x) uniform across wave -> uniform branches
    float acc = dw_b[lane];
#pragma unroll
    for (int ky = 0; ky < K_; ++ky) {
        int yy = y + ky - PAD_;
        if (yy < 0 || yy >= H_) continue;
#pragma unroll
        for (int kx = 0; kx < K_; ++kx) {
            int xx = x + kx - PAD_;
            if (xx < 0 || xx >= W_) continue;
            acc += qc[yy * W_ + xx] * dw_w[lane * (K_ * K_) + ky * K_ + kx];
        }
    }

    // LayerNorm over the 32 channels of this wave
    float s = acc;
    for (int m = 16; m; m >>= 1) s += __shfl_xor(s, m, 32);
    float mu = s * (1.0f / GC_);
    float d  = acc - mu;
    float v  = d * d;
    for (int m = 16; m; m >>= 1) v += __shfl_xor(v, m, 32);
    float var = v * (1.0f / GC_);
    float yln = d * rsqrtf(var + EPS_) * ln_w[lane] + ln_b[lane];

    // exact GELU
    float gel = 0.5f * yln * (1.0f + erff(yln * 0.70710678118654752f));

    // pointwise 32 -> 2 (dy, dx)
    float py = gel * pw_w[lane];
    float px = gel * pw_w[GC_ + lane];
    for (int m = 16; m; m >>= 1) {
        py += __shfl_xor(py, m, 32);
        px += __shfl_xor(px, m, 32);
    }
    float offy = tanhf(py) * (1.0f / (H_ - 1));
    float offx = tanhf(px) * (1.0f / (W_ - 1));

    float ry = ((0.5f + (float)y) / (float)(H_ - 1)) * 2.0f - 1.0f;
    float rx = ((0.5f + (float)x) / (float)(W_ - 1)) * 2.0f - 1.0f;
    float gy = offy + ry, gx = offx + rx;

    float iy = (gy + 1.0f) * 0.5f * (float)(H_ - 1);
    float ix = (gx + 1.0f) * 0.5f * (float)(W_ - 1);
    float x0f = floorf(ix), y0f = floorf(iy);
    float wx1 = ix - x0f, wy1 = iy - y0f;
    int x0 = (int)x0f, y0 = (int)y0f;

    const float* img = X + ((size_t)b * C_ + cg) * HW_;
    float out = 0.f;
#pragma unroll
    for (int cy = 0; cy < 2; ++cy) {
#pragma unroll
        for (int cx = 0; cx < 2; ++cx) {
            int xi = x0 + cx, yi = y0 + cy;
            bool inb = (xi >= 0) && (xi <= W_ - 1) && (yi >= 0) && (yi <= H_ - 1);
            int xc = xi < 0 ? 0 : (xi > W_ - 1 ? W_ - 1 : xi);
            int yc = yi < 0 ? 0 : (yi > H_ - 1 ? H_ - 1 : yi);
            float wgt = (cx ? wx1 : 1.f - wx1) * (cy ? wy1 : 1.f - wy1);
            out += img[yc * W_ + xc] * (inb ? wgt : 0.f);
        }
    }
    Xs[((size_t)b * C_ + cg) * HW_ + hw] = out;
}

// ===================== depthwise 3x3 LEPE ============================
__global__ __launch_bounds__(256)
void lepe3x3(const float* __restrict__ Q, const float* __restrict__ w,
             const float* __restrict__ bias, float* __restrict__ Out) {
    int t = blockIdx.x * blockDim.x + threadIdx.x;
    if (t >= B_ * C_ * HW_) return;
    int hw = t % HW_;
    int c  = (t / HW_) % C_;
    int y = hw / W_, x = hw % W_;
    const float* qc = Q + (size_t)(t - hw);   // base of [b, c] plane
    float acc = bias[c];
#pragma unroll
    for (int ky = 0; ky < 3; ++ky) {
        int yy = y + ky - 1;
        if (yy < 0 || yy >= H_) continue;
#pragma unroll
        for (int kx = 0; kx < 3; ++kx) {
            int xx = x + kx - 1;
            if (xx < 0 || xx >= W_) continue;
            acc += qc[yy * W_ + xx] * w[c * 9 + ky * 3 + kx];
        }
    }
    Out[t] = acc;
}

// =====================================================================
// Neighborhood attention. One wave per (b, head, pixel); lane = head dim.
// Each logit: per-lane product + 5-step butterfly -> value in ALL lanes,
// so softmax over 49 logits is done redundantly per lane in registers.
// Padded neighbors: dot = 0 but RPB still enters softmax; V contributes 0.
// Adds LEPE residual on store.
// =====================================================================
__global__ __launch_bounds__(256)
void attn_kernel(const float* __restrict__ Q, const float* __restrict__ Km,
                 const float* __restrict__ Vm, const float* __restrict__ rpb,
                 const float* __restrict__ lep, float* __restrict__ Out) {
    const int lane = threadIdx.x & 31;
    const int wave = blockIdx.x * 8 + (threadIdx.x >> 5);
    if (wave >= B_ * HEADS_ * HW_) return;
    const int hw = wave % HW_;
    const int bh = wave / HW_;
    const int h  = bh % HEADS_;
    const int b  = bh / HEADS_;
    const int y = hw / W_, x = hw % W_;
    const size_t chan = (size_t)b * C_ * HW_ + (size_t)(h * HD_ + lane) * HW_;

    const float qd = Q[chan + hw] * SCALE_;

    float logit[P_];
#pragma unroll
    for (int p = 0; p < P_; ++p) {
        int dy = p / K_ - PAD_, dx = p % K_ - PAD_;
        int ny = y + dy, nx = x + dx;
        bool inb = (ny >= 0) && (ny < H_) && (nx >= 0) && (nx < W_);
        float kv = inb ? Km[chan + ny * W_ + nx] : 0.f;
        float prod = qd * kv;
        for (int m = 16; m; m >>= 1) prod += __shfl_xor(prod, m, 32);
        // rpb [4,13,13], crop [0:7, 0:7], dy-major
        logit[p] = prod + rpb[(h * 13 + (3 + dy)) * 13 + (3 + dx)];
    }

    float mx = logit[0];
#pragma unroll
    for (int p = 1; p < P_; ++p) mx = fmaxf(mx, logit[p]);
    float ssum = 0.f;
#pragma unroll
    for (int p = 0; p < P_; ++p) {
        logit[p] = __expf(logit[p] - mx);
        ssum += logit[p];
    }
    float inv = 1.0f / ssum;

    float outd = 0.f;
#pragma unroll
    for (int p = 0; p < P_; ++p) {
        int dy = p / K_ - PAD_, dx = p % K_ - PAD_;
        int ny = y + dy, nx = x + dx;
        bool inb = (ny >= 0) && (ny < H_) && (nx >= 0) && (nx < W_);
        float vv = inb ? Vm[chan + ny * W_ + nx] : 0.f;
        outd += logit[p] * inv * vv;
    }
    Out[chan + hw] = outd + lep[chan + hw];
}

// =====================================================================
extern "C" void kernel_launch(void* const* d_in, const int* in_sizes, int n_in,
                              void* d_out, int out_size, void* d_ws, size_t ws_size,
                              hipStream_t stream) {
    const float* x    = (const float*)d_in[0];
    const float* Wq   = (const float*)d_in[1];
    const float* bq   = (const float*)d_in[2];
    const float* Wk   = (const float*)d_in[3];
    const float* bk   = (const float*)d_in[4];
    const float* Wv   = (const float*)d_in[5];
    const float* bv   = (const float*)d_in[6];
    const float* Wo   = (const float*)d_in[7];
    const float* bo   = (const float*)d_in[8];
    const float* dww  = (const float*)d_in[9];
    const float* dwb  = (const float*)d_in[10];
    const float* lnw  = (const float*)d_in[11];
    const float* lnb  = (const float*)d_in[12];
    const float* pww  = (const float*)d_in[13];
    const float* rpew = (const float*)d_in[14];
    const float* rpeb = (const float*)d_in[15];
    const float* rpb  = (const float*)d_in[16];

    float* ws = (float*)d_ws;
    const size_t SZ = (size_t)B_ * C_ * HW_;     // 1,605,632 floats = 6.4 MB
    float* q    = ws;
    float* xs   = ws + SZ;
    float* kmap = ws + 2 * SZ;
    float* vmap = ws + 3 * SZ;
    float* lepw = ws + 4 * SZ;
    float* aout = xs;                            // reuse: x_sampled dead after K/V GEMMs

    const int gemmBlocks = (NPIX / 16) / 4;                       // 196 (4 waves/blk)
    const int waveUnits  = B_ * HEADS_ * HW_;                     // 50176
    const int waveBlocks = (waveUnits + 7) / 8;                   // 6272
    const int eltBlocks  = (B_ * C_ * HW_ + 255) / 256;           // 6272

    gemm_conv1x1_wmma<<<gemmBlocks, dim3(128), 0, stream>>>(x, Wq, bq, q);
    offset_sample<<<waveBlocks, dim3(256), 0, stream>>>(x, q, dww, dwb, lnw, lnb, pww, xs);
    lepe3x3<<<eltBlocks, dim3(256), 0, stream>>>(q, rpew, rpeb, lepw);
    gemm_conv1x1_wmma<<<gemmBlocks, dim3(128), 0, stream>>>(xs, Wk, bk, kmap);
    gemm_conv1x1_wmma<<<gemmBlocks, dim3(128), 0, stream>>>(xs, Wv, bv, vmap);
    attn_kernel<<<waveBlocks, dim3(256), 0, stream>>>(q, kmap, vmap, rpb, lepw, aout);
    gemm_conv1x1_wmma<<<gemmBlocks, dim3(128), 0, stream>>>(aout, Wo, bo, (float*)d_out);
}
